// Attention_3049426780809
// MI455X (gfx1250) — matrix operand order repaired and checked by reading, not compiled, not run on hardware
//
#include <hip/hip_runtime.h>

// ---------------------------------------------------------------------------
// Types
// ---------------------------------------------------------------------------
typedef __bf16 bf16;
typedef bf16  bf16x16 __attribute__((ext_vector_type(16)));
typedef float f32x8   __attribute__((ext_vector_type(8)));
typedef int   v4i     __attribute__((vector_size(16)));

union Frag16 {
    bf16x16 v;
    uint4   q[2];
};

__device__ __forceinline__ f32x8 wmma_bf16(bf16x16 a, bf16x16 b, f32x8 c) {
    // D = A(16x32) * B(32x16) + C(16x16 f32) -> v_wmma_f32_16x16x32_bf16
    return __builtin_amdgcn_wmma_f32_16x16x32_bf16(
        false, a, false, b, (short)0, c, false, false);
}

// A-fragment (16x32 bf16), source row-major, leading dim `ld` (elements).
// lane l -> row m0+(l&15); elems 0..7 = K kb..kb+7, 8..15 = K 16+kb.., kb=(l>=16)?8:0.
__device__ __forceinline__ bf16x16
load_a_frag(const bf16* __restrict__ base, int ld, int m0, int k0, int lane) {
    const bf16* p = base + (size_t)(m0 + (lane & 15)) * ld + k0 + ((lane >> 4) * 8);
    Frag16 f;
    f.q[0] = *(const uint4*)(p);
    f.q[1] = *(const uint4*)(p + 16);
    return f.v;
}

// B-fragment (32x16 bf16) from a K-contiguous ("transposed") source src[n][k].
// lane l -> col n0+(l&15); elems 0..7 = K kb..kb+7, 8..15 = K 16+kb.., kb=(l>=16)?8:0.
__device__ __forceinline__ bf16x16
load_b_fragT(const bf16* __restrict__ srcT, int ld, int n0, int k0, int lane) {
    const bf16* p = srcT + (size_t)(n0 + (lane & 15)) * ld + k0 + ((lane >> 4) * 8);
    Frag16 f;
    f.q[0] = *(const uint4*)(p);
    f.q[1] = *(const uint4*)(p + 16);
    return f.v;
}

// ---------------------------------------------------------------------------
// Feature detection
// ---------------------------------------------------------------------------
#if defined(__has_builtin)
#if __has_builtin(__builtin_amdgcn_update_dpp)
#define HAVE_DPP 1
#endif
#if __has_builtin(__builtin_amdgcn_global_load_async_to_lds_b128)
#define HAVE_ASYNC_LDS 1
#endif
#endif

// ---------------------------------------------------------------------------
// Row-of-16 all-reduce via DPP (v_mov_b32_dpp row_ror:N), VALU only.
// ---------------------------------------------------------------------------
#ifdef HAVE_DPP
template <int CTRL>
__device__ __forceinline__ float dpp_rot(float v) {
    return __int_as_float(
        __builtin_amdgcn_update_dpp(0, __float_as_int(v), CTRL, 0xf, 0xf, true));
}
#endif

__device__ __forceinline__ float row16_max(float v) {
#ifdef HAVE_DPP
    v = fmaxf(v, dpp_rot<0x121>(v));   // row_ror:1
    v = fmaxf(v, dpp_rot<0x122>(v));   // row_ror:2
    v = fmaxf(v, dpp_rot<0x124>(v));   // row_ror:4
    v = fmaxf(v, dpp_rot<0x128>(v));   // row_ror:8
#else
#pragma unroll
    for (int d = 1; d < 16; d <<= 1) v = fmaxf(v, __shfl_xor(v, d));
#endif
    return v;
}

__device__ __forceinline__ float row16_sum(float v) {
#ifdef HAVE_DPP
    v += dpp_rot<0x121>(v);
    v += dpp_rot<0x122>(v);
    v += dpp_rot<0x124>(v);
    v += dpp_rot<0x128>(v);
#else
#pragma unroll
    for (int d = 1; d < 16; d <<= 1) v += __shfl_xor(v, d);
#endif
    return v;
}

// ---------------------------------------------------------------------------
// 16B global -> LDS copy: async DMA path (ASYNCcnt, global_load_async_to_lds_b128)
// when available, otherwise VGPR round-trip.
// ---------------------------------------------------------------------------
__device__ __forceinline__ void copy16_g2l(bf16* lds_dst, const bf16* gsrc) {
#ifdef HAVE_ASYNC_LDS
    __builtin_amdgcn_global_load_async_to_lds_b128(
        (__attribute__((address_space(1))) v4i*)(const void*)gsrc,
        (__attribute__((address_space(3))) v4i*)(void*)lds_dst,
        0, 0);
#else
    *(uint4*)lds_dst = *(const uint4*)gsrc;
#endif
}

__device__ __forceinline__ void wait_g2l() {
#ifdef HAVE_ASYNC_LDS
#if __has_builtin(__builtin_amdgcn_s_wait_asynccnt)
    __builtin_amdgcn_s_wait_asynccnt(0);
#else
    asm volatile("s_wait_asynccnt 0x0" ::: "memory");
#endif
#endif
}

// ---------------------------------------------------------------------------
// f32 -> bf16 conversion (plain)
// ---------------------------------------------------------------------------
__global__ void cvt_f32_bf16(const float* __restrict__ src, bf16* __restrict__ dst, int n) {
    int i = blockIdx.x * blockDim.x + threadIdx.x;
    if (i < n) dst[i] = (bf16)src[i];
}

// f32 [K,N] -> bf16 [N,K] (weights, one-time). 32x32 tiles via LDS.
__global__ __launch_bounds__(256)
void cvt_transpose_f32_bf16(const float* __restrict__ src, bf16* __restrict__ dst,
                            int K, int N) {
    __shared__ float tile[32][33];
    const int k0 = blockIdx.y * 32, n0 = blockIdx.x * 32;
    const int tx = threadIdx.x & 31, ty = threadIdx.x >> 5;   // ty 0..7
#pragma unroll
    for (int r = ty; r < 32; r += 8)
        tile[r][tx] = src[(size_t)(k0 + r) * N + n0 + tx];
    __syncthreads();
#pragma unroll
    for (int r = ty; r < 32; r += 8)
        dst[(size_t)(n0 + r) * K + k0 + tx] = (bf16)tile[tx][r];
}

// ---------------------------------------------------------------------------
// bf16 GEMM: C[M,N] = A[M,K] * Bt[N,K]^T  (+bias for f32 output)
// 128x128 block tile, BK=32, 256 threads = 8 waves, wave tile 32x64.
// LDS leading dim padded to 40 elements -> conflict-free b128 fragment loads.
// ---------------------------------------------------------------------------
#define LDK 40

template <bool OUT_BF16>
__global__ __launch_bounds__(256)
void gemm_bf16_kernel(const bf16* __restrict__ A, const bf16* __restrict__ Bt,
                      const float* __restrict__ bias, void* __restrict__ Cv,
                      int M, int N, int K) {
    __shared__ __align__(16) bf16 As[128 * LDK];   // [m][k]
    __shared__ __align__(16) bf16 Bs[128 * LDK];   // [n][k]

    const int tid  = threadIdx.x;
    const int lane = tid & 31;
    const int wave = tid >> 5;            // 0..7
    const int wm   = (wave >> 1) * 32;    // 0,32,64,96
    const int wn   = (wave & 1) * 64;     // 0,64
    const int m0   = blockIdx.y * 128;
    const int n0   = blockIdx.x * 128;

    f32x8 acc[2][4];
#pragma unroll
    for (int i = 0; i < 2; ++i)
#pragma unroll
        for (int j = 0; j < 4; ++j) { f32x8 z = {}; acc[i][j] = z; }

    for (int k0 = 0; k0 < K; k0 += 32) {
        // --- stage A (128x32) and Bt (128x32) tiles (async DMA to LDS) ---
#pragma unroll
        for (int c = tid; c < 512; c += 256) {
            int row = c >> 2, col = (c & 3) * 8;
            copy16_g2l(&As[row * LDK + col], &A[(size_t)(m0 + row) * K + k0 + col]);
            copy16_g2l(&Bs[row * LDK + col], &Bt[(size_t)(n0 + row) * K + k0 + col]);
        }
        wait_g2l();
        __syncthreads();

        if (k0 + 32 < K) {   // hint next K tile into cache (global_prefetch_b8)
            __builtin_prefetch(&A[(size_t)(m0 + (tid >> 1)) * K + k0 + 32], 0, 0);
            __builtin_prefetch(&Bt[(size_t)(n0 + (tid >> 1)) * K + k0 + 32], 0, 0);
        }

        bf16x16 af0 = load_a_frag(As, LDK, wm,      0, lane);
        bf16x16 af1 = load_a_frag(As, LDK, wm + 16, 0, lane);
#pragma unroll
        for (int j = 0; j < 4; ++j) {
            bf16x16 bfj = load_b_fragT(Bs, LDK, wn + j * 16, 0, lane);
            acc[0][j] = wmma_bf16(af0, bfj, acc[0][j]);
            acc[1][j] = wmma_bf16(af1, bfj, acc[1][j]);
        }
        __syncthreads();
    }

    // --- epilogue: C/D layout lane l -> row r+8*(l>>4), col (l&15) ---
    const int colb = lane & 15;
    const int rowh = (lane >> 4) * 8;
#pragma unroll
    for (int i = 0; i < 2; ++i) {
#pragma unroll
        for (int j = 0; j < 4; ++j) {
            int gn = n0 + wn + j * 16 + colb;
#pragma unroll
            for (int r = 0; r < 8; ++r) {
                int gm = m0 + wm + i * 16 + rowh + r;
                float v = acc[i][j][r];
                if constexpr (OUT_BF16) {
                    ((bf16*)Cv)[(size_t)gm * N + gn] = (bf16)v;
                } else {
                    ((float*)Cv)[(size_t)gm * N + gn] = v + bias[gn];
                }
            }
        }
    }
}

// ---------------------------------------------------------------------------
// Flash attention: one block = one (b,h) x 64 query rows, 4 waves x 16 rows.
// 64-key chunks. qkv: bf16 [8192,3072], col layout (qkv_idx*1024 + h*64 + d).
// ---------------------------------------------------------------------------
#define VLD 72    // padded leading dim for Vs/Ps (conflict-free b128 gathers)

__global__ __launch_bounds__(128)
void attn_kernel(const bf16* __restrict__ qkv, bf16* __restrict__ outp) {
    __shared__ __align__(16) bf16 Vs[64 * VLD];      // [d][key] transposed chunk
    __shared__ __align__(16) bf16 Ps[4][16 * VLD];   // per-wave P tile [row][key]

    const int tid  = threadIdx.x;
    const int lane = tid & 31;
    const int wave = tid >> 5;
    const int bh   = blockIdx.x >> 5;     // 0..63
    const int rblk = blockIdx.x & 31;     // 0..31
    const int b    = bh >> 4, h = bh & 15;
    const size_t rowbase = (size_t)b * 2048;
    const int row0 = rblk * 64 + wave * 16;

    const bf16* Qp = qkv + (rowbase + row0) * 3072 + h * 64;
    const bf16* Kp = qkv + rowbase * 3072 + 1024 + h * 64;
    const bf16* Vp = qkv + rowbase * 3072 + 2048 + h * 64;

    // Q tile 16x64 kept in registers as two A-frags (d-chunks 0..31, 32..63)
    bf16x16 qf0 = load_a_frag(Qp, 3072, 0, 0,  lane);
    bf16x16 qf1 = load_a_frag(Qp, 3072, 0, 32, lane);

    f32x8 of[4];
#pragma unroll
    for (int j = 0; j < 4; ++j) { f32x8 z = {}; of[j] = z; }
    float mi[8], li[8];
#pragma unroll
    for (int r = 0; r < 8; ++r) { mi[r] = -1e30f; li[r] = 0.0f; }

    const float scale = 0.125f;   // 64^-0.5

    for (int kc = 0; kc < 2048; kc += 64) {
        __syncthreads();   // all waves done reading previous Vs
        // cooperative stage: V chunk [64 keys][64 d] -> Vs[d][key]
#pragma unroll
        for (int c = tid; c < 512; c += 128) {
            int row = c >> 3;            // key within chunk
            int col = (c & 7) * 8;       // d segment
            union { uint4 q; bf16 h8[8]; } t;
            t.q = *(const uint4*)&Vp[(size_t)(kc + row) * 3072 + col];
#pragma unroll
            for (int i = 0; i < 8; ++i) Vs[(size_t)(col + i) * VLD + row] = t.h8[i];
        }
        __syncthreads();

        // S = Q K^T : four 16-key subtiles, contraction over d (2 WMMA steps)
        f32x8 s[4];
#pragma unroll
        for (int j = 0; j < 4; ++j) {
            f32x8 c = {};
            bf16x16 bk0 = load_b_fragT(Kp, 3072, kc + j * 16, 0,  lane);
            c = wmma_bf16(qf0, bk0, c);
            bf16x16 bk1 = load_b_fragT(Kp, 3072, kc + j * 16, 32, lane);
            c = wmma_bf16(qf1, bk1, c);
            s[j] = c;
        }

        // online softmax update (row stats replicated across each 16-lane half)
        bf16* pb = &Ps[wave][0];
#pragma unroll
        for (int r = 0; r < 8; ++r) {
            float v0 = s[0][r] * scale;
            float v1 = s[1][r] * scale;
            float v2 = s[2][r] * scale;
            float v3 = s[3][r] * scale;
            float vmax  = row16_max(fmaxf(fmaxf(v0, v1), fmaxf(v2, v3)));
            float mnew  = fmaxf(mi[r], vmax);
            float alpha = __expf(mi[r] - mnew);
            float p0 = __expf(v0 - mnew);
            float p1 = __expf(v1 - mnew);
            float p2 = __expf(v2 - mnew);
            float p3 = __expf(v3 - mnew);
            float rsum = row16_sum((p0 + p1) + (p2 + p3));
            li[r] = li[r] * alpha + rsum;
            mi[r] = mnew;
#pragma unroll
            for (int j = 0; j < 4; ++j) of[j][r] *= alpha;
            // C-layout -> LDS (row-major [16][VLD]) for A-frag reshape
            int prow = r + ((lane >> 4) * 8);
            bf16* pr = pb + prow * VLD + (lane & 15);
            pr[0]  = (bf16)p0;
            pr[16] = (bf16)p1;
            pr[32] = (bf16)p2;
            pr[48] = (bf16)p3;
        }

        // O += P * V  (contraction over 64 keys = 2 WMMA steps)
        bf16x16 pf0 = load_a_frag(pb, VLD, 0, 0,  lane);
        bf16x16 pf1 = load_a_frag(pb, VLD, 0, 32, lane);
#pragma unroll
        for (int j = 0; j < 4; ++j) {
            bf16x16 vf0 = load_b_fragT(Vs, VLD, j * 16, 0,  lane);
            of[j] = wmma_bf16(pf0, vf0, of[j]);
            bf16x16 vf1 = load_b_fragT(Vs, VLD, j * 16, 32, lane);
            of[j] = wmma_bf16(pf1, vf1, of[j]);
        }
    }

    // epilogue: out[b, n, h*64 + d] = O / l
    const int colb = lane & 15;
    const int rowh = (lane >> 4) * 8;
#pragma unroll
    for (int r = 0; r < 8; ++r) {
        float inv = 1.0f / li[r];
        size_t gm = rowbase + row0 + rowh + r;
#pragma unroll
        for (int j = 0; j < 4; ++j) {
            outp[gm * 1024 + h * 64 + j * 16 + colb] = (bf16)(of[j][r] * inv);
        }
    }
}

// ---------------------------------------------------------------------------
// Launch
// ---------------------------------------------------------------------------
extern "C" void kernel_launch(void* const* d_in, const int* in_sizes, int n_in,
                              void* d_out, int out_size, void* d_ws, size_t ws_size,
                              hipStream_t stream) {
    const float* inp  = (const float*)d_in[0];   // [4,2048,1024]
    const float* Wqkv = (const float*)d_in[1];   // [1024,3072]
    const float* Wmlp = (const float*)d_in[2];   // [1024,1024]
    const float* bmlp = (const float*)d_in[3];   // [1024]
    float* out = (float*)d_out;                  // [4,2048,1024]

    const int M = 8192, DIM = 1024, N3 = 3072;

    char* ws = (char*)d_ws;
    size_t o = 0;
    bf16* Xb    = (bf16*)(ws + o); o += (size_t)M   * DIM * 2;   // 16.8 MB
    bf16* WqkvT = (bf16*)(ws + o); o += (size_t)DIM * N3  * 2;   //  6.3 MB  [3072,1024]
    bf16* WmlpT = (bf16*)(ws + o); o += (size_t)DIM * DIM * 2;   //  2.1 MB  [1024,1024]
    bf16* QKVb  = (bf16*)(ws + o); o += (size_t)M   * N3  * 2;   // 50.3 MB
    bf16* Attb  = (bf16*)(ws + o);                                // 16.8 MB

    cvt_f32_bf16<<<(M * DIM + 255) / 256, 256, 0, stream>>>(inp, Xb, M * DIM);
    cvt_transpose_f32_bf16<<<dim3(N3 / 32, DIM / 32), 256, 0, stream>>>(Wqkv, WqkvT, DIM, N3);
    cvt_transpose_f32_bf16<<<dim3(DIM / 32, DIM / 32), 256, 0, stream>>>(Wmlp, WmlpT, DIM, DIM);

    // QKV projection: [8192,1024] x [1024,3072] -> bf16 [8192,3072]
    gemm_bf16_kernel<true><<<dim3(N3 / 128, M / 128), 256, 0, stream>>>(
        Xb, WqkvT, nullptr, (void*)QKVb, M, N3, DIM);

    // Flash attention: 64 heads x 32 row-blocks
    attn_kernel<<<2048, 128, 0, stream>>>(QKVb, Attb);

    // MLP: [8192,1024] x [1024,1024] + bias -> f32 out
    gemm_bf16_kernel<false><<<dim3(DIM / 128, M / 128), 256, 0, stream>>>(
        Attb, WmlpT, bmlp, (void*)out, M, DIM, DIM);
}
